// GCN_12807592476806
// MI455X (gfx1250) — matrix operand order repaired and checked by reading, not compiled, not run on hardware
//
#include <hip/hip_runtime.h>
#include <stdint.h>

#define HID 128

typedef __attribute__((ext_vector_type(16))) __bf16   v16bf;
typedef __attribute__((ext_vector_type(8)))  float    v8f;
typedef __attribute__((ext_vector_type(4)))  unsigned u32x4;

// ---------- bf16 split helpers ----------
__device__ __forceinline__ unsigned rnd_bf16(float f) {
    unsigned x = __float_as_uint(f);
    return (x + 0x7FFFu + ((x >> 16) & 1u)) >> 16;
}
__device__ __forceinline__ void atomic_addf(float* p, float v) {
    __hip_atomic_fetch_add(p, v, __ATOMIC_RELAXED, __HIP_MEMORY_SCOPE_AGENT);
}

// ---------- utility kernels ----------
__global__ void k_zero(float4* p, long n4) {
    long i = (long)blockIdx.x * blockDim.x + threadIdx.x;
    if (i < n4) p[i] = make_float4(0.f, 0.f, 0.f, 0.f);
}

// degrees for both graphs: deg[0]=in_gt, deg[1]=out_gt, deg[2]=in_at, deg[3]=out_at
__global__ void k_deg(const int* gs, const int* gd, const int* as_, const int* ad,
                      float* deg, int E, int N) {
    int i = blockIdx.x * blockDim.x + threadIdx.x;
    if (i >= E) return;
    atomic_addf(&deg[0 * N + gd[i]], 1.0f);
    atomic_addf(&deg[1 * N + gs[i]], 1.0f);
    atomic_addf(&deg[2 * N + ad[i]], 1.0f);
    atomic_addf(&deg[3 * N + as_[i]], 1.0f);
}

// per-node scalars: scal[g*3+0]=1/(in+1), [1]=norm_src(out), [2]=norm_dst(in)
__global__ void k_norm(const float* deg, float* scal, int N) {
    int n = blockIdx.x * blockDim.x + threadIdx.x;
    if (n >= N) return;
    for (int g = 0; g < 2; ++g) {
        float din  = deg[(2 * g + 0) * N + n];
        float dout = deg[(2 * g + 1) * N + n];
        scal[(3 * g + 0) * (long)N + n] = 1.0f / (din + 1.0f);
        scal[(3 * g + 1) * (long)N + n] = (dout > 0.f) ? rsqrtf(fmaxf(dout, 1.f)) : 0.f;
        scal[(3 * g + 2) * (long)N + n] = (din  > 0.f) ? rsqrtf(fmaxf(din,  1.f)) : 0.f;
    }
}

// transpose + split weight into hi/lo bf16 planes: Wh/Wl[n*K+k] = split(W[k][n])
__global__ void k_packw(const float* __restrict__ W, unsigned short* __restrict__ Wh,
                        unsigned short* __restrict__ Wl, int K, int NC) {
    int i = blockIdx.x * blockDim.x + threadIdx.x;
    if (i >= K * NC) return;
    int n = i / K, k = i - n * K;
    float f = W[k * NC + n];
    unsigned hi = rnd_bf16(f);
    float fh = __uint_as_float(hi << 16);
    unsigned lo = rnd_bf16(f - fh);
    Wh[i] = (unsigned short)hi;
    Wl[i] = (unsigned short)lo;
}

// edge scatter-sum: out[dst] += in[src], row width ld (2^chshift threads/edge, 4 floats each)
__global__ void k_scatter(const float* __restrict__ in, const int* __restrict__ src,
                          const int* __restrict__ dst, float* __restrict__ out,
                          int E, int ld, int chshift) {
    long gid = (long)blockIdx.x * blockDim.x + threadIdx.x;
    long e = gid >> chshift;
    if (e >= E) return;
    int c = (int)(gid & ((1 << chshift) - 1)) << 2;
    const float4 v = *(const float4*)(in + (long)src[e] * ld + c);
    float* o = out + (long)dst[e] * ld + c;
    atomic_addf(o + 0, v.x);
    atomic_addf(o + 1, v.y);
    atomic_addf(o + 2, v.z);
    atomic_addf(o + 3, v.w);
}

// ---------- WMMA GEMM: out[N x NC] = epi(((A+Aadd)*ascale) @ W + bias) ----------
// K = 128. COLT = NC/16 column tiles. 256 threads = 8 waves. Split-bf16 hi/lo planes,
// f32 accumulate: ah*bh + ah*bl + al*bh (~fp32 accuracy). Operands are formed by
// contiguous 16B loads (LDS swizzle for A; [n][k] planes for B) -> no repacking.
template <int COLT>
__global__ __launch_bounds__(256) void k_gemm(
    const float* __restrict__ A, const float* __restrict__ Aadd,
    const float* __restrict__ ascale,
    const unsigned short* __restrict__ Wh, const unsigned short* __restrict__ Wl,
    const float* __restrict__ bias, const float* __restrict__ oscale,
    float* __restrict__ out, int N, int relu) {
    constexpr int ROWS_PB = 16 * (8 / COLT);
    constexpr int NC = COLT * 16;
    __shared__ unsigned short sAh[ROWS_PB * HID];
    __shared__ unsigned short sAl[ROWS_PB * HID];

    const int tid = threadIdx.x;
    const int blockrow0 = blockIdx.x * ROWS_PB;

    // Stage A tile (fused SAGE pre-op), split into hi/lo planes with WMMA swizzle:
    // pos = row*128 + ((k>>4)&1)*64 + (k>>5)*16 + (k&15)
    for (int i = tid; i < ROWS_PB * HID; i += 256) {
        int rl = i >> 7, k = i & 127;
        int gr = blockrow0 + rl;
        float f = 0.f;
        if (gr < N) {
            f = A[(long)gr * HID + k];
            if (Aadd)   f += Aadd[(long)gr * HID + k];
            if (ascale) f *= ascale[gr];
        }
        unsigned hi = rnd_bf16(f);
        float fh = __uint_as_float(hi << 16);
        unsigned lo = rnd_bf16(f - fh);
        int sw = (rl << 7) + (((k >> 4) & 1) << 6) + ((k >> 5) << 4) + (k & 15);
        sAh[sw] = (unsigned short)hi;
        sAl[sw] = (unsigned short)lo;
    }
    __syncthreads();

    const int wave = tid >> 5, lane = tid & 31;
    const int hf = lane >> 4, l15 = lane & 15;
    const int colg = (wave % COLT) * 16;
    const int rtile = (wave / COLT) * 16;
    const int rowbase = blockrow0 + rtile;
    if (rowbase >= N) return;

    float cb = bias ? bias[colg + l15] : 0.f;
    v8f c;
#pragma unroll
    for (int r = 0; r < 8; ++r) c[r] = cb;

    const int arow = rtile + l15;
    const unsigned short* aph = sAh + arow * HID + hf * 64;
    const unsigned short* apl = sAl + arow * HID + hf * 64;
    const unsigned short* bph = Wh + (long)(colg + l15) * HID + hf * 16;
    const unsigned short* bpl = Wl + (long)(colg + l15) * HID + hf * 16;

    union V16 { v16bf v; u32x4 q[2]; };

#pragma unroll
    for (int kb = 0; kb < HID; kb += 32) {
        V16 ah, al, bh, bl;
        const int ao = kb >> 1;  // contiguous 16 ushorts per lane after swizzle
        ah.q[0] = *(const u32x4*)(aph + ao);
        ah.q[1] = *(const u32x4*)(aph + ao + 8);
        al.q[0] = *(const u32x4*)(apl + ao);
        al.q[1] = *(const u32x4*)(apl + ao + 8);
        bh.q[0] = *(const u32x4*)(bph + kb);
        bh.q[1] = *(const u32x4*)(bph + kb + 8);
        bl.q[0] = *(const u32x4*)(bpl + kb);
        bl.q[1] = *(const u32x4*)(bpl + kb + 8);
        c = __builtin_amdgcn_wmma_f32_16x16x32_bf16(false, ah.v, false, bh.v, (short)0, c, false, false);
        c = __builtin_amdgcn_wmma_f32_16x16x32_bf16(false, ah.v, false, bl.v, (short)0, c, false, false);
        c = __builtin_amdgcn_wmma_f32_16x16x32_bf16(false, al.v, false, bh.v, (short)0, c, false, false);
    }

    // C/D layout: element (m = r + hf*8, n = l15)
#pragma unroll
    for (int r = 0; r < 8; ++r) {
        int gr = rowbase + r + hf * 8;
        if (gr < N) {
            float o = c[r];
            if (relu) o = fmaxf(o, 0.f);
            if (oscale) o *= oscale[gr];
            out[(long)gr * NC + colg + l15] = o;
        }
    }
}

// xcat[:, head*16 + j] = agg16 * norm_dst + b3
__global__ void k_headout(const float* __restrict__ agg16, const float* __restrict__ nd,
                          const float* __restrict__ b3, float* __restrict__ xcat,
                          int head, int N) {
    int i = blockIdx.x * blockDim.x + threadIdx.x;
    if (i >= N * 16) return;
    int n = i >> 4, j = i & 15;
    xcat[(long)n * HID + head * 16 + j] = agg16[i] * nd[n] + b3[j];
}

// out = base + LayerNorm(xin)*g + b ; one wave32 per 128-float row
__global__ __launch_bounds__(256) void k_ln(const float* __restrict__ base,
                                            const float* __restrict__ xin,
                                            const float* __restrict__ g,
                                            const float* __restrict__ b,
                                            float* __restrict__ out, int N) {
    int wave = threadIdx.x >> 5, lane = threadIdx.x & 31;
    long row = (long)blockIdx.x * 8 + wave;
    if (row >= N) return;
    float4 x = ((const float4*)(xin + row * HID))[lane];
    float s = x.x + x.y + x.z + x.w;
#pragma unroll
    for (int off = 16; off; off >>= 1) s += __shfl_xor(s, off, 32);
    float mu = s * (1.0f / 128.0f);
    float4 d = make_float4(x.x - mu, x.y - mu, x.z - mu, x.w - mu);
    float v = d.x * d.x + d.y * d.y + d.z * d.z + d.w * d.w;
#pragma unroll
    for (int off = 16; off; off >>= 1) v += __shfl_xor(v, off, 32);
    float rs = rsqrtf(v * (1.0f / 128.0f) + 1e-5f);
    float4 gg = ((const float4*)g)[lane];
    float4 bb = ((const float4*)b)[lane];
    float4 ba = ((const float4*)(base + row * HID))[lane];
    float4 o;
    o.x = ba.x + d.x * rs * gg.x + bb.x;
    o.y = ba.y + d.y * rs * gg.y + bb.y;
    o.z = ba.z + d.z * rs * gg.z + bb.z;
    o.w = ba.w + d.w * rs * gg.w + bb.w;
    ((float4*)(out + row * HID))[lane] = o;
}

// ---------- host launcher ----------
static inline int cdiv(long a, long b) { return (int)((a + b - 1) / b); }

extern "C" void kernel_launch(void* const* d_in, const int* in_sizes, int n_in,
                              void* d_out, int out_size, void* d_ws, size_t ws_size,
                              hipStream_t stream) {
    const float* h       = (const float*)d_in[0];
    const int*   gt_src  = (const int*)d_in[1];
    const int*   gt_dst  = (const int*)d_in[2];
    const int*   at_src  = (const int*)d_in[3];
    const int*   at_dst  = (const int*)d_in[4];
    const float* W1      = (const float*)d_in[5];
    const float* b1      = (const float*)d_in[6];
    const float* W2      = (const float*)d_in[7];
    const float* b2      = (const float*)d_in[8];
    const float* W3      = (const float*)d_in[9];
    const float* b3      = (const float*)d_in[10];
    const float* ffn_W1  = (const float*)d_in[11];
    const float* ffn_b1  = (const float*)d_in[12];
    const float* ffn_W2  = (const float*)d_in[13];
    const float* ffn_b2  = (const float*)d_in[14];
    const float* ln_g    = (const float*)d_in[15];
    const float* ln_b    = (const float*)d_in[16];

    const int N = in_sizes[0] / HID;
    const int E = in_sizes[1];
    float* ws = (float*)d_ws;
    float* out = (float*)d_out;

    // workspace layout (floats)
    long off = 0;
    float* deg   = ws + off; off += 4L * N;
    float* scal  = ws + off; off += 6L * N;   // [inv1,ns,nd] x {gt, at}
    float* agg   = ws + off; off += (long)N * HID;
    float* t1    = ws + off; off += (long)N * HID;
    float* t2    = ws + off; off += (long)N * HID;
    float* xcat  = ws + off; off += (long)N * HID;
    float* x1    = ws + off; off += (long)N * HID;
    float* feat  = ws + off; off += 16L * N;
    float* agg16 = ws + off; off += 16L * N;
    // bf16 hi/lo weight planes (ushort); sizes in floats = elems/2
    const long szW12 = 8L * HID * HID, szW3 = 8L * HID * 16, szF = (long)HID * HID;
    unsigned short* W1h = (unsigned short*)(ws + off); off += szW12 / 2;
    unsigned short* W1l = (unsigned short*)(ws + off); off += szW12 / 2;
    unsigned short* W2h = (unsigned short*)(ws + off); off += szW12 / 2;
    unsigned short* W2l = (unsigned short*)(ws + off); off += szW12 / 2;
    unsigned short* W3h = (unsigned short*)(ws + off); off += szW3 / 2;
    unsigned short* W3l = (unsigned short*)(ws + off); off += szW3 / 2;
    unsigned short* F1h = (unsigned short*)(ws + off); off += szF / 2;
    unsigned short* F1l = (unsigned short*)(ws + off); off += szF / 2;
    unsigned short* F2h = (unsigned short*)(ws + off); off += szF / 2;
    unsigned short* F2l = (unsigned short*)(ws + off); off += szF / 2;

    const dim3 B(256);
    // degrees + per-node scalars
    k_zero<<<cdiv(4L * N / 4, 256), B, 0, stream>>>((float4*)deg, 4L * N / 4);
    k_deg<<<cdiv(E, 256), B, 0, stream>>>(gt_src, gt_dst, at_src, at_dst, deg, E, N);
    k_norm<<<cdiv(N, 256), B, 0, stream>>>(deg, scal, N);

    // pre-pack weights (transpose + bf16 hi/lo planes)
    for (int i = 0; i < 8; ++i) {
        k_packw<<<cdiv(HID * HID, 256), B, 0, stream>>>(W1 + (long)i * HID * HID,
            W1h + (long)i * HID * HID, W1l + (long)i * HID * HID, HID, HID);
        k_packw<<<cdiv(HID * HID, 256), B, 0, stream>>>(W2 + (long)i * HID * HID,
            W2h + (long)i * HID * HID, W2l + (long)i * HID * HID, HID, HID);
        k_packw<<<cdiv(HID * 16, 256),  B, 0, stream>>>(W3 + (long)i * HID * 16,
            W3h + (long)i * HID * 16,  W3l + (long)i * HID * 16,  HID, 16);
    }
    k_packw<<<cdiv(HID * HID, 256), B, 0, stream>>>(ffn_W1, F1h, F1l, HID, HID);
    k_packw<<<cdiv(HID * HID, 256), B, 0, stream>>>(ffn_W2, F2h, F2l, HID, HID);

    const int gemm8_grid = cdiv(N, 16);
    const int gemm1_grid = cdiv(N, 128);
    const long n128_4 = (long)N * HID / 4, n16_4 = 16L * N / 4;

    for (int head = 0; head < 8; ++head) {
        const bool is_at = ((head >> 1) & 1) != 0;  // heads: gt,gt,at,at,gt,gt,at,at
        const int* src = is_at ? at_src : gt_src;
        const int* dst = is_at ? at_dst : gt_dst;
        const float* inv1 = scal + (is_at ? 3L : 0L) * N;
        const float* ns   = inv1 + N;
        const float* nd   = inv1 + 2L * N;

        // SAGE 1: agg = scatter(h); t1 = ((agg+h)*inv1) @ W1 + b1
        k_zero<<<cdiv(n128_4, 256), B, 0, stream>>>((float4*)agg, n128_4);
        k_scatter<<<cdiv((long)E * 32, 256), B, 0, stream>>>(h, src, dst, agg, E, HID, 5);
        k_gemm<8><<<gemm8_grid, B, 0, stream>>>(agg, h, inv1,
            W1h + (long)head * HID * HID, W1l + (long)head * HID * HID,
            b1 + (long)head * HID, nullptr, t1, N, 0);
        // SAGE 2
        k_zero<<<cdiv(n128_4, 256), B, 0, stream>>>((float4*)agg, n128_4);
        k_scatter<<<cdiv((long)E * 32, 256), B, 0, stream>>>(t1, src, dst, agg, E, HID, 5);
        k_gemm<8><<<gemm8_grid, B, 0, stream>>>(agg, t1, inv1,
            W2h + (long)head * HID * HID, W2l + (long)head * HID * HID,
            b2 + (long)head * HID, nullptr, t2, N, 0);
        // GraphConv: feat = (t2 @ W3) * ns ; agg16 = scatter(feat); slice = agg16*nd + b3
        k_gemm<1><<<gemm1_grid, B, 0, stream>>>(t2, nullptr, nullptr,
            W3h + (long)head * HID * 16, W3l + (long)head * HID * 16,
            nullptr, ns, feat, N, 0);
        k_zero<<<cdiv(n16_4, 256), B, 0, stream>>>((float4*)agg16, n16_4);
        k_scatter<<<cdiv((long)E * 4, 256), B, 0, stream>>>(feat, src, dst, agg16, E, 16, 2);
        k_headout<<<cdiv((long)N * 16, 256), B, 0, stream>>>(agg16, nd, b3 + head * 16, xcat, head, N);
    }

    // x1 = h + LN(xcat); ffn = relu(x1@F1+b)@F2+b; out = x1 + LN(ffn)
    k_ln<<<cdiv(N, 8), B, 0, stream>>>(h, xcat, ln_g, ln_b, x1, N);
    k_gemm<8><<<gemm8_grid, B, 0, stream>>>(x1, nullptr, nullptr, F1h, F1l, ffn_b1, nullptr, agg, N, 1);
    k_gemm<8><<<gemm8_grid, B, 0, stream>>>(agg, nullptr, nullptr, F2h, F2l, ffn_b2, nullptr, t1, N, 0);
    k_ln<<<cdiv(N, 8), B, 0, stream>>>(x1, t1, ln_g, ln_b, out, N);
    (void)n_in; (void)out_size; (void)ws_size;
}
